// GraphConvLayer_90323162235537
// MI455X (gfx1250) — matrix operand order repaired and checked by reading, not compiled, not run on hardware
//
#include <hip/hip_runtime.h>

// GraphConv: out = relu((A @ X) @ W + bias),  B=4, N=4096, D=F=128.
// HBM-bound on A (268 MB @ 23.3 TB/s ~= 11.5 us) -> stream A once, convert
// f32->bf16 in registers, v_wmma_f32_16x16x32_bf16 with f32 accumulation.
// LDS double-buffered, single barrier per K-tile, fused epilogue GEMM.

#define BN 4096
#define DD 128
#define FF 128
#define KT 32
#define NITER (BN / KT)

typedef __bf16 bf16;
typedef __attribute__((ext_vector_type(16))) __bf16 v16bf;
typedef __attribute__((ext_vector_type(8)))  float  v8f;
typedef __attribute__((ext_vector_type(4)))  unsigned int v4u;
typedef __attribute__((ext_vector_type(4)))  __bf16 v4bf;

#define SA_STRIDE 40            // 32 + 8 pad bf16 -> 80B rows (16B aligned)
#define SX_STRIDE 40
#define SW_STRIDE 136           // 128 + 8 pad -> 272B rows (16B aligned)
#define SG_STRIDE 136
#define STAGE_ELEMS (128 * SA_STRIDE + 128 * SX_STRIDE)   // 10240 bf16 per buffer

// Two contiguous 16B LDS chunks -> one 16x32 bf16 WMMA operand (per-lane half).
__device__ __forceinline__ v16bf load_frag16(const bf16* p0, const bf16* p1) {
    union { v16bf v; v4u q[2]; } u;
    u.q[0] = *(const v4u*)p0;
    u.q[1] = *(const v4u*)p1;
    return u.v;
}

__device__ __forceinline__ v4bf cvt4(float4 f) {
    v4bf p; p.x = (bf16)f.x; p.y = (bf16)f.y; p.z = (bf16)f.z; p.w = (bf16)f.w;
    return p;
}

__global__ __launch_bounds__(256)
void graphconv_wmma_kernel(const float* __restrict__ A,
                           const float* __restrict__ X,
                           const float* __restrict__ W,
                           const float* __restrict__ bias,
                           float* __restrict__ out)
{
    __shared__ __align__(16) bf16  sStage[2][STAGE_ELEMS]; // [A tile | X^T tile], x2
    __shared__ __align__(16) bf16  sWT[128 * SW_STRIDE];   // weight [f][d] bf16
    __shared__ float sBias[FF];

    const int tid  = threadIdx.x;
    const int wave = tid >> 5;
    const int lane = tid & 31;
    const int half = lane >> 4;
    const int l16  = lane & 15;

    const int b  = blockIdx.x >> 5;            // 32 row-tiles per batch
    const int m0 = (blockIdx.x & 31) * 128;

    const float* Ab = A + (size_t)b * BN * BN + (size_t)m0 * BN;
    const float* Xb = X + (size_t)b * BN * DD;

    // ---- one-time: W -> sWT[f][d] bf16 via 4x4 register micro-transpose ----
    #pragma unroll
    for (int j = 0; j < 4; ++j) {
        int bb   = tid + j * 256;              // 1024 4x4 blocks
        int dblk = (bb >> 5) << 2;
        int fblk = (bb & 31) << 2;
        float4 r0 = *(const float4*)(W + (size_t)(dblk + 0) * FF + fblk);
        float4 r1 = *(const float4*)(W + (size_t)(dblk + 1) * FF + fblk);
        float4 r2 = *(const float4*)(W + (size_t)(dblk + 2) * FF + fblk);
        float4 r3 = *(const float4*)(W + (size_t)(dblk + 3) * FF + fblk);
        v4bf c0 = {(bf16)r0.x, (bf16)r1.x, (bf16)r2.x, (bf16)r3.x};
        v4bf c1 = {(bf16)r0.y, (bf16)r1.y, (bf16)r2.y, (bf16)r3.y};
        v4bf c2 = {(bf16)r0.z, (bf16)r1.z, (bf16)r2.z, (bf16)r3.z};
        v4bf c3 = {(bf16)r0.w, (bf16)r1.w, (bf16)r2.w, (bf16)r3.w};
        *(v4bf*)(sWT + (fblk + 0) * SW_STRIDE + dblk) = c0;
        *(v4bf*)(sWT + (fblk + 1) * SW_STRIDE + dblk) = c1;
        *(v4bf*)(sWT + (fblk + 2) * SW_STRIDE + dblk) = c2;
        *(v4bf*)(sWT + (fblk + 3) * SW_STRIDE + dblk) = c3;
    }
    if (tid < FF) sBias[tid] = bias[tid];

    // X staging roles: each wave owns 4 k-rows, each lane a 4-wide d column group.
    const int kbq  = wave << 2;                // k sub-block base (0,4,...,28)
    const int dcol = lane << 2;                // d column base (0..124)

    float4 ra[4], rx[4];

    auto load_tiles = [&](int kk) {
        const int k0 = kk * KT;
        #pragma unroll
        for (int i = 0; i < 4; ++i) {          // A tile: 1024 float4, coalesced rows
            int q  = tid + i * 256;
            int r  = q >> 3;
            int c4 = (q & 7) << 2;
            const float* gp = Ab + (size_t)r * BN + k0 + c4;
            ra[i] = *(const float4*)gp;
            __builtin_prefetch(gp + KT, 0, 1); // pull next tile's line into L2
        }
        #pragma unroll
        for (int j = 0; j < 4; ++j)            // X tile: 512B contiguous per wave
            rx[j] = *(const float4*)(Xb + (size_t)(k0 + kbq + j) * DD + dcol);
    };

    auto store_tiles = [&](int p) {
        bf16* sA  = &sStage[p][0];
        bf16* sXT = &sStage[p][128 * SA_STRIDE];
        #pragma unroll
        for (int i = 0; i < 4; ++i) {
            int q  = tid + i * 256;
            int r  = q >> 3;
            int c4 = (q & 7) << 2;
            *(v4bf*)(sA + r * SA_STRIDE + c4) = cvt4(ra[i]);
        }
        // 4x4 micro-transpose: packed 8B column stores instead of 16x b16
        v4bf c0 = {(bf16)rx[0].x, (bf16)rx[1].x, (bf16)rx[2].x, (bf16)rx[3].x};
        v4bf c1 = {(bf16)rx[0].y, (bf16)rx[1].y, (bf16)rx[2].y, (bf16)rx[3].y};
        v4bf c2 = {(bf16)rx[0].z, (bf16)rx[1].z, (bf16)rx[2].z, (bf16)rx[3].z};
        v4bf c3 = {(bf16)rx[0].w, (bf16)rx[1].w, (bf16)rx[2].w, (bf16)rx[3].w};
        *(v4bf*)(sXT + (dcol + 0) * SX_STRIDE + kbq) = c0;
        *(v4bf*)(sXT + (dcol + 1) * SX_STRIDE + kbq) = c1;
        *(v4bf*)(sXT + (dcol + 2) * SX_STRIDE + kbq) = c2;
        *(v4bf*)(sXT + (dcol + 3) * SX_STRIDE + kbq) = c3;
    };

    v8f acc[8] = {};                           // 16(M) x 128(D) f32 strip per wave
    const int mrow = (wave << 4) + l16;

    // ========== Stage 1: agg = A @ X, pipelined, 1 barrier / K-tile ==========
    load_tiles(0);
    for (int kk = 0; kk < NITER; ++kk) {
        const int p = kk & 1;
        store_tiles(p);
        if (kk + 1 < NITER) load_tiles(kk + 1);   // overlap with compute below
        __syncthreads();

        const bf16* sA  = &sStage[p][0];
        const bf16* sXT = &sStage[p][128 * SA_STRIDE];
        const bf16* ap  = sA + mrow * SA_STRIDE + half * 8;
        const v16bf afrag = load_frag16(ap, ap + 16);
        #pragma unroll
        for (int nt = 0; nt < 8; ++nt) {
            const bf16* bp = sXT + (nt * 16 + l16) * SX_STRIDE + half * 8;
            const v16bf bfrag = load_frag16(bp, bp + 16);
            acc[nt] = __builtin_amdgcn_wmma_f32_16x16x32_bf16(
                false, afrag, false, bfrag, (short)0, acc[nt], false, false);
        }
        // no trailing barrier: next iteration writes the other buffer
    }
    __syncthreads();                            // before reusing stage LDS as agg

    // ========== spill agg -> bf16 LDS (re-layout C/D -> A fragments) =========
    bf16* sAgg = &sStage[0][0];                 // 17408 elems needed, 20480 avail
    #pragma unroll
    for (int nt = 0; nt < 8; ++nt) {
        #pragma unroll
        for (int r = 0; r < 8; ++r)             // C/D: VGPR r -> M = r + 8*half
            sAgg[(wave * 16 + r + half * 8) * SG_STRIDE + nt * 16 + l16] =
                (bf16)acc[nt][r];
    }
    __syncthreads();

    // ========== Stage 2: out = relu(agg @ W + bias), K = 128 =================
    v8f acc2[8] = {};
    #pragma unroll
    for (int kc = 0; kc < 4; ++kc) {
        const int kb = kc * 32;
        const bf16* ap = sAgg + mrow * SG_STRIDE + kb + half * 8;
        const v16bf afrag = load_frag16(ap, ap + 16);
        #pragma unroll
        for (int nt = 0; nt < 8; ++nt) {
            const bf16* bp = sWT + (nt * 16 + l16) * SW_STRIDE + kb + half * 8;
            const v16bf bfrag = load_frag16(bp, bp + 16);
            acc2[nt] = __builtin_amdgcn_wmma_f32_16x16x32_bf16(
                false, afrag, false, bfrag, (short)0, acc2[nt], false, false);
        }
    }

    // bias + relu + f32 store
    float* outb = out + ((size_t)b * BN + m0) * FF;
    #pragma unroll
    for (int nt = 0; nt < 8; ++nt) {
        const int f = nt * 16 + l16;
        const float bv = sBias[f];
        #pragma unroll
        for (int r = 0; r < 8; ++r) {
            const int m = wave * 16 + r + half * 8;
            float v = acc2[nt][r] + bv;
            outb[(size_t)m * FF + f] = v > 0.0f ? v : 0.0f;
        }
    }
}

extern "C" void kernel_launch(void* const* d_in, const int* in_sizes, int n_in,
                              void* d_out, int out_size, void* d_ws, size_t ws_size,
                              hipStream_t stream) {
    // setup_inputs order: features [B,N,D], A [B,N,N], weight [D,F], bias [F]
    const float* features = (const float*)d_in[0];
    const float* A        = (const float*)d_in[1];
    const float* weight   = (const float*)d_in[2];
    const float* bias     = (const float*)d_in[3];
    float* out = (float*)d_out;

    dim3 grid(4 * (BN / 128));   // B * row-tiles = 128 workgroups
    dim3 block(256);             // 8 waves (wave32)
    hipLaunchKernelGGL(graphconv_wmma_kernel, grid, block, 0, stream,
                       A, features, weight, bias, out);
}
